// DeformableTransformerFusionLayerV2_41446434407090
// MI455X (gfx1250) — matrix operand order repaired and checked by reading, not compiled, not run on hardware
//
#include <hip/hip_runtime.h>
#include <math.h>

typedef __attribute__((ext_vector_type(16))) _Float16 v16h;
typedef __attribute__((ext_vector_type(8)))  float    v8f;

#define TM 128
#define TN 64
#define TK 32
#define LDS_STRIDE (TK + 8)   // halfs; pad to dodge bank conflicts (80B row)

union AFrag { v16h v; _Float16 h[16]; };
union CFrag { v8f  v; float    f[8];  };
union H4    { _Float16 h[4]; unsigned long long u; };

// ---------------------------------------------------------------------------
// Generic fused GEMM: C[M,N] = A[M,K] @ W[K,N] + bias (+ optional exact GELU)
// Block = 256 threads = 8 waves. Block tile 128x64, K-step 32, f16 WMMA.
// A rows past M are *clamped* (not zeroed): a garbage A-tile row m only feeds
// output row m, which the epilogue masks — so the K-loop stays branchless.
// ---------------------------------------------------------------------------
__global__ __launch_bounds__(256) void gemm_bias_kernel(
    const float* __restrict__ A, const float* __restrict__ W,
    const float* __restrict__ bias, float* __restrict__ C,
    int M, int N, int K, int gelu)
{
  __shared__ _Float16 lA[TM * LDS_STRIDE];
  __shared__ _Float16 lB[TN * LDS_STRIDE];

  const int tid  = threadIdx.x;
  const int wave = tid >> 5;
  const int lane = tid & 31;
  const int bm   = blockIdx.y * TM;
  const int bn   = blockIdx.x * TN;

  CFrag acc[4];
  #pragma unroll
  for (int s = 0; s < 4; ++s)
    #pragma unroll
    for (int r = 0; r < 8; ++r) acc[s].f[r] = 0.0f;

  for (int k0 = 0; k0 < K; k0 += TK) {
    // ---- stage A tile (TM x TK), f32 -> f16, packed b64 stores -------------
    #pragma unroll
    for (int i = 0; i < 4; ++i) {
      int e   = tid * 16 + i * 4;        // 256 thr * 16 elems = 4096 = 128*32
      int row = e >> 5;
      int col = e & 31;
      int grow = bm + row; if (grow > M - 1) grow = M - 1;   // branchless clamp
      const float* gp = A + (size_t)grow * K + k0 + col;
      float4 val = *(const float4*)gp;
      if (i == 0 && k0 + TK < K) __builtin_prefetch(gp + TK, 0, 3);
      H4 p;
      p.h[0] = (_Float16)val.x; p.h[1] = (_Float16)val.y;
      p.h[2] = (_Float16)val.z; p.h[3] = (_Float16)val.w;
      *(unsigned long long*)&lA[row * LDS_STRIDE + col] = p.u;
    }
    // ---- stage B tile (TK x TN) -> lB[n][k] via 4x4 register transpose -----
    if (tid < 128) {
      int kk = (tid >> 4) * 4;           // 0..28
      int nn = (tid & 15) * 4;           // 0..60
      float va[4][4];
      #pragma unroll
      for (int j = 0; j < 4; ++j) {
        const float* gp = W + (size_t)(k0 + kk + j) * N + bn + nn;
        float4 v = *(const float4*)gp;
        va[j][0] = v.x; va[j][1] = v.y; va[j][2] = v.z; va[j][3] = v.w;
        if (j == 0 && k0 + TK < K) __builtin_prefetch(gp + (size_t)TK * N, 0, 3);
      }
      #pragma unroll
      for (int c = 0; c < 4; ++c) {
        H4 p;
        p.h[0] = (_Float16)va[0][c]; p.h[1] = (_Float16)va[1][c];
        p.h[2] = (_Float16)va[2][c]; p.h[3] = (_Float16)va[3][c];
        *(unsigned long long*)&lB[(nn + c) * LDS_STRIDE + kk] = p.u;
      }
    }
    __syncthreads();

    // ---- A fragment (ISA 16-bit A 16x32 layout) ----------------------------
    AFrag a;
    {
      int r  = lane & 15;
      int kb = (lane >> 4) * 8;
      const _Float16* arow = &lA[(wave * 16 + r) * LDS_STRIDE];
      #pragma unroll
      for (int v = 0; v < 8; ++v) {
        int k = (v < 4) ? (kb + 2 * v) : (16 + kb + 2 * (v - 4));
        a.h[2 * v]     = arow[k];
        a.h[2 * v + 1] = arow[k + 1];
      }
    }
    // ---- 4 x WMMA, reusing the A fragment ----------------------------------
    #pragma unroll
    for (int s = 0; s < 4; ++s) {
      AFrag b;
      int n   = s * 16 + (lane & 15);
      int kb2 = (lane >> 4) * 8;
      const _Float16* brow = &lB[n * LDS_STRIDE];
      #pragma unroll
      for (int v = 0; v < 8; ++v) {
        int k = (v < 4) ? (kb2 + 2 * v) : (16 + kb2 + 2 * (v - 4));
        b.h[2 * v]     = brow[k];
        b.h[2 * v + 1] = brow[k + 1];
      }
      acc[s].v = __builtin_amdgcn_wmma_f32_16x16x32_f16(
          false, a.v, false, b.v, (short)0, acc[s].v, false, false);
    }
    __syncthreads();
  }

  // ---- epilogue: bias (+GELU), ISA C/D layout ------------------------------
  #pragma unroll
  for (int s = 0; s < 4; ++s) {
    int col  = bn + s * 16 + (lane & 15);
    float bv = bias ? bias[col] : 0.0f;
    #pragma unroll
    for (int r = 0; r < 8; ++r) {
      int row = bm + wave * 16 + (lane >> 4) * 8 + r;
      if (row < M) {
        float x = acc[s].f[r] + bv;
        if (gelu) x = 0.5f * x * (1.0f + erff(x * 0.70710678118654752f));
        C[(size_t)row * N + col] = x;
      }
    }
  }
}

// ---------------------------------------------------------------------------
// LayerNorm over rows of 256: one wave per row, optional residual add.
// ---------------------------------------------------------------------------
__global__ void layernorm_kernel(const float* __restrict__ x,
                                 const float* __restrict__ res,
                                 const float* __restrict__ g,
                                 const float* __restrict__ b,
                                 float* __restrict__ y, int rows)
{
  int wave = (int)((blockIdx.x * blockDim.x + threadIdx.x) >> 5);
  int lane = threadIdx.x & 31;
  if (wave >= rows) return;
  const float* xr = x + (size_t)wave * 256;
  const float* rr = res ? res + (size_t)wave * 256 : nullptr;
  float v[8];
  float sum = 0.f;
  #pragma unroll
  for (int j = 0; j < 8; ++j) {
    int c = lane + j * 32;
    float t = xr[c];
    if (rr) t += rr[c];
    v[j] = t; sum += t;
  }
  #pragma unroll
  for (int o = 16; o > 0; o >>= 1) sum += __shfl_xor(sum, o, 32);
  float mean = sum * (1.0f / 256.0f);
  float var = 0.f;
  #pragma unroll
  for (int j = 0; j < 8; ++j) { float d = v[j] - mean; var += d * d; }
  #pragma unroll
  for (int o = 16; o > 0; o >>= 1) var += __shfl_xor(var, o, 32);
  float inv = rsqrtf(var * (1.0f / 256.0f) + 1e-5f);
  #pragma unroll
  for (int j = 0; j < 8; ++j) {
    int c = lane + j * 32;
    y[(size_t)wave * 256 + c] = (v[j] - mean) * inv * g[c] + b[c];
  }
}

// ---------------------------------------------------------------------------
// Softmax over groups of 16 (per token x head attention weights), in place.
// ---------------------------------------------------------------------------
__global__ void softmax16_kernel(float* __restrict__ aw, int groups)
{
  int gi = blockIdx.x * blockDim.x + threadIdx.x;
  if (gi >= groups) return;
  float* p = aw + (size_t)gi * 16;
  float v[16], m = -1e30f;
  #pragma unroll
  for (int i = 0; i < 16; ++i) { v[i] = p[i]; m = fmaxf(m, v[i]); }
  float s = 0.f;
  #pragma unroll
  for (int i = 0; i < 16; ++i) { v[i] = expf(v[i] - m); s += v[i]; }
  float r = 1.0f / s;
  #pragma unroll
  for (int i = 0; i < 16; ++i) p[i] = v[i] * r;
}

// ---------------------------------------------------------------------------
// q = tgt + query_pos
// ---------------------------------------------------------------------------
__global__ void add_kernel(const float* __restrict__ a, const float* __restrict__ b,
                           float* __restrict__ c, size_t n)
{
  size_t i = (size_t)blockIdx.x * blockDim.x + threadIdx.x;
  if (i < n) c[i] = a[i] + b[i];
}

// ---------------------------------------------------------------------------
// Deformable sampling: one wave per (token, head); lane = channel (0..31).
// Each bilinear corner gather is one coalesced 32-lane b32 load.
// ---------------------------------------------------------------------------
__global__ void sample_kernel(const float* __restrict__ value,
                              const float* __restrict__ off,
                              const float* __restrict__ aw,
                              const float* __restrict__ ref,
                              float* __restrict__ out, int B, int Lq)
{
  const int LH[4] = {92, 46, 23, 12};
  const int LW[4] = {92, 46, 23, 12};
  const int LS[4] = {0, 8464, 10580, 11109};

  int unit = blockIdx.x * (blockDim.x >> 5) + (threadIdx.x >> 5);
  int lane = threadIdx.x & 31;
  int total = B * Lq * 8;
  if (unit >= total) return;
  int head = unit & 7;
  int tok  = unit >> 3;
  int b    = tok / Lq;
  int Lv   = Lq;

  float acc = 0.f;
  #pragma unroll
  for (int l = 0; l < 4; ++l) {
    int H = LH[l], W = LW[l], st = LS[l];
    float refx = ref[((size_t)tok * 4 + l) * 2 + 0];
    float refy = ref[((size_t)tok * 4 + l) * 2 + 1];
    const float* vbase = value + ((size_t)b * Lv + st) * 256 + head * 32 + lane;
    #pragma unroll
    for (int p = 0; p < 4; ++p) {
      size_t ob = (size_t)tok * 256 + head * 32 + l * 8 + p * 2;
      float ox = off[ob], oy = off[ob + 1];
      float w  = aw[(size_t)tok * 128 + head * 16 + l * 4 + p];
      float x = (refx + ox / (float)W) * (float)W - 0.5f;
      float y = (refy + oy / (float)H) * (float)H - 0.5f;
      int x0 = (int)floorf(x), y0 = (int)floorf(y);
      #pragma unroll
      for (int dy = 0; dy < 2; ++dy)
        #pragma unroll
        for (int dx = 0; dx < 2; ++dx) {
          int xi = x0 + dx, yi = y0 + dy;
          if (xi >= 0 && xi < W && yi >= 0 && yi < H) {
            float wc = (1.f - fabsf(x - (float)xi)) * (1.f - fabsf(y - (float)yi));
            acc += w * wc * vbase[(size_t)(yi * W + xi) * 256];
          }
        }
    }
  }
  out[(size_t)tok * 256 + head * 32 + lane] = acc;
}

// ---------------------------------------------------------------------------
extern "C" void kernel_launch(void* const* d_in, const int* in_sizes, int n_in,
                              void* d_out, int out_size, void* d_ws, size_t ws_size,
                              hipStream_t stream)
{
  const float* tgt    = (const float*)d_in[0];
  const float* qpos   = (const float*)d_in[1];
  const float* ref    = (const float*)d_in[2];
  const float* src    = (const float*)d_in[3];
  const float* w_ds   = (const float*)d_in[6];
  const float* b_ds   = (const float*)d_in[7];
  const float* g_ds   = (const float*)d_in[8];
  const float* be_ds  = (const float*)d_in[9];
  const float* w_off  = (const float*)d_in[10];
  const float* b_off  = (const float*)d_in[11];
  const float* w_attn = (const float*)d_in[12];
  const float* b_attn = (const float*)d_in[13];
  const float* w_val  = (const float*)d_in[14];
  const float* b_val  = (const float*)d_in[15];
  const float* w_out  = (const float*)d_in[16];
  const float* b_out  = (const float*)d_in[17];
  const float* w_cs   = (const float*)d_in[18];
  const float* b_cs   = (const float*)d_in[19];
  const float* g1     = (const float*)d_in[20];
  const float* be1    = (const float*)d_in[21];
  const float* w1     = (const float*)d_in[22];
  const float* b1     = (const float*)d_in[23];
  const float* g2     = (const float*)d_in[24];
  const float* be2    = (const float*)d_in[25];
  float* out = (float*)d_out;

  const int NT = in_sizes[0] / 256;   // B * Lq
  const int Lq = 11253;
  const int B  = NT / Lq;

  float* q     = (float*)d_ws;
  float* s     = q     + (size_t)NT * 256;
  float* value = s     + (size_t)NT * 256;
  float* offb  = value + (size_t)NT * 256;
  float* awb   = offb  + (size_t)NT * 256;
  float* attn  = awb   + (size_t)NT * 128;
  float* tmp   = attn  + (size_t)NT * 256;
  float* t     = tmp   + (size_t)NT * 256;

  const int mtiles = (NT + TM - 1) / TM;
  dim3 blk(256);
  dim3 grid256(256 / TN, mtiles);
  dim3 grid128(128 / TN, mtiles);
  int lnBlocks = (NT + 7) / 8;                 // 8 rows per 256-thread block
  size_t nelem = (size_t)NT * 256;

  // s = LN(src @ w_ds + b_ds)
  gemm_bias_kernel<<<grid256, blk, 0, stream>>>(src, w_ds, b_ds, tmp, NT, 256, 256, 0);
  layernorm_kernel<<<lnBlocks, 256, 0, stream>>>(tmp, nullptr, g_ds, be_ds, s, NT);
  // value = s @ w_val + b_val
  gemm_bias_kernel<<<grid256, blk, 0, stream>>>(s, w_val, b_val, value, NT, 256, 256, 0);
  // q = tgt + query_pos
  add_kernel<<<(int)((nelem + 255) / 256), 256, 0, stream>>>(tgt, qpos, q, nelem);
  // off = q @ w_off + b_off ; aw = softmax(q @ w_attn + b_attn)
  gemm_bias_kernel<<<grid256, blk, 0, stream>>>(q, w_off, b_off, offb, NT, 256, 256, 0);
  gemm_bias_kernel<<<grid128, blk, 0, stream>>>(q, w_attn, b_attn, awb, NT, 128, 256, 0);
  softmax16_kernel<<<(NT * 8 + 255) / 256, 256, 0, stream>>>(awb, NT * 8);
  // deformable sampling -> attn
  sample_kernel<<<(NT * 8 + 7) / 8, 256, 0, stream>>>(value, offb, awb, ref, attn, B, Lq);
  // t2 = (attn @ w_out + b_out) @ w_cs + b_cs ; t = LN(tgt + t2)
  gemm_bias_kernel<<<grid256, blk, 0, stream>>>(attn, w_out, b_out, tmp, NT, 256, 256, 0);
  gemm_bias_kernel<<<grid256, blk, 0, stream>>>(tmp, w_cs, b_cs, s, NT, 256, 256, 0);
  layernorm_kernel<<<lnBlocks, 256, 0, stream>>>(s, tgt, g1, be1, t, NT);
  // h = gelu(t @ w1 + b1) ; out = LN(t + h)
  gemm_bias_kernel<<<grid256, blk, 0, stream>>>(t, w1, b1, tmp, NT, 256, 256, 1);
  layernorm_kernel<<<lnBlocks, 256, 0, stream>>>(tmp, t, g2, be2, out, NT);
}
